// LSTMEncoderDecoderWrapper_26912265077468
// MI455X (gfx1250) — compile-verified, matrix operand order checked by reading, not run on hardware
//
#include <hip/hip_runtime.h>
#include <math.h>

#define B_    64
#define S_    512
#define F_    64
#define H_    512
#define G4_   2048      // 4*H
#define U_    128
#define TOUT_ 96
#define KCAT_ 576       // H + F (fused [Whh | Wih] K dim)
#define NWG_  32
#define KCH_  96        // K-chunk staged in LDS (6 chunks of 96 = 576)

typedef __attribute__((ext_vector_type(16))) __bf16 v16bf;
typedef __attribute__((ext_vector_type(8)))  float  v8f;
typedef __attribute__((ext_vector_type(4)))  unsigned v4u;
typedef __attribute__((ext_vector_type(8)))  int      v8i;
typedef __attribute__((ext_vector_type(4)))  int      v4i;

#if __has_builtin(__builtin_amdgcn_tensor_load_to_lds) && \
    __has_builtin(__builtin_amdgcn_s_wait_tensorcnt)
#define USE_TDM 1
#else
#define USE_TDM 0
#endif

union Frag { v16bf v; uint4 q[2]; };

__device__ __forceinline__ unsigned short f2bf(float f) {
  unsigned u = __builtin_bit_cast(unsigned, f);
  return (unsigned short)((u + 0x7FFFu + ((u >> 16) & 1u)) >> 16);  // RNE
}
__device__ __forceinline__ float sigf(float x) { return 1.0f / (1.0f + __expf(-x)); }

// Load a 16x32 bf16 WMMA fragment (A or B operand) from a row-major matrix.
// Lanes 0-15 hold row r, K {0..7,16..23}; lanes 16-31 K {8..15,24..31}.
__device__ __forceinline__ v16bf load_frag(const unsigned short* __restrict__ m,
                                           int ld, int r0, int k0) {
  const int lane = threadIdx.x & 31;
  const int r = lane & 15, hf = lane >> 4;
  const unsigned short* p = m + (size_t)(r0 + r) * ld + (k0 + hf * 8);
  Frag f;
  f.q[0] = *(const uint4*)(p);
  f.q[1] = *(const uint4*)(p + 16);
  return f.v;
}

__device__ __forceinline__ v8f wmma_bf16(v16bf a, v16bf b, v8f c) {
  return __builtin_amdgcn_wmma_f32_16x16x32_bf16(false, a, false, b, (short)0, c,
                                                 false, false);
}

// Device-wide sense barrier (counter + generation), agent scope.
__device__ __forceinline__ void gsync(unsigned* cnt, unsigned* gen) {
  __syncthreads();
  if (threadIdx.x == 0) {
    __threadfence();
    unsigned g = __hip_atomic_load(gen, __ATOMIC_RELAXED, __HIP_MEMORY_SCOPE_AGENT);
    unsigned a = __hip_atomic_fetch_add(cnt, 1u, __ATOMIC_ACQ_REL, __HIP_MEMORY_SCOPE_AGENT);
    if (a == NWG_ - 1) {
      __hip_atomic_store(cnt, 0u, __ATOMIC_RELAXED, __HIP_MEMORY_SCOPE_AGENT);
      __hip_atomic_fetch_add(gen, 1u, __ATOMIC_ACQ_REL, __HIP_MEMORY_SCOPE_AGENT);
    } else {
      while (__hip_atomic_load(gen, __ATOMIC_ACQUIRE, __HIP_MEMORY_SCOPE_AGENT) == g)
        __builtin_amdgcn_s_sleep(2);
    }
  }
  __syncthreads();
}

#if USE_TDM
// Issue one TDM 2D tile load: 64 rows x KCH_ bf16 cols from a row-major
// [64 x KCAT_] bf16 tensor into a compact [64 x KCH_] LDS tile.
// D# layout per CDNA5 ISA.
__device__ __forceinline__ void tdm_stage(const unsigned short* gsrc, unsigned lds_off) {
  unsigned long long ga = (unsigned long long)(size_t)gsrc;
  v4u g0;
  g0[0] = 1u;                                             // count=1, user D#
  g0[1] = lds_off;                                        // lds_addr
  g0[2] = (unsigned)(ga & 0xFFFFFFFFu);                   // global_addr[31:0]
  g0[3] = (unsigned)((ga >> 32) & 0x01FFFFFFu) | 0x80000000u;  // addr[56:32] | type=2
  v8i g1;
  g1[0] = (int)(1u << 16);                                // data_size=1 (2 bytes)
  g1[1] = (int)((unsigned)KCAT_ << 16);                   // tensor_dim0[15:0] @bit48
  g1[2] = (int)(((unsigned)KCAT_ >> 16) | (64u << 16));   // dim0 hi | tensor_dim1=64
  g1[3] = (int)((unsigned)KCH_ << 16);                    // tile_dim0=KCH_
  g1[4] = 64;                                             // tile_dim1=64, tile_dim2=0
  g1[5] = KCAT_;                                          // tensor_dim0_stride lo32
  g1[6] = 0;
  g1[7] = 0;
  v4i z4 = {0, 0, 0, 0};
  v8i z8 = {0, 0, 0, 0, 0, 0, 0, 0};
  __builtin_amdgcn_tensor_load_to_lds(g0, g1, z4, z4, z8, 0);
}
#endif

// Stage K-chunk kc of the activation matrix (64 x KCAT_ bf16) into LDS tile.
__device__ __forceinline__ void stage_chunk(unsigned short (*buf)[KCH_],
                                            const unsigned short* hxr,
                                            int kc, int tid) {
#if USE_TDM
  if ((tid >> 5) == 0)
    tdm_stage(hxr + kc * KCH_, (unsigned)(size_t)&buf[0][0]);
#else
  for (int i = tid; i < 64 * KCH_ / 8; i += 256) {
    int row = i / (KCH_ / 8), cq = i % (KCH_ / 8);
    *(uint4*)&buf[row][cq * 8] =
        *(const uint4*)&hxr[(size_t)row * KCAT_ + kc * KCH_ + cq * 8];
  }
#endif
}

__device__ __forceinline__ void stage_wait(int tid) {
#if USE_TDM
  if ((tid >> 5) == 0) __builtin_amdgcn_s_wait_tensorcnt(0);
#endif
  __syncthreads();
}

// ---- one-time weight conversion / bias folding ----
__global__ __launch_bounds__(256) void prep_kernel(
    const float* __restrict__ enc_Wih, const float* __restrict__ enc_Whh,
    const float* __restrict__ enc_bih, const float* __restrict__ enc_bhh,
    const float* __restrict__ att_b1,  const float* __restrict__ att_b2,
    const float* __restrict__ att_W2,  const float* __restrict__ dec_Wih,
    const float* __restrict__ dec_bih, const float* __restrict__ dec_bhh,
    unsigned short* __restrict__ Wenc, unsigned short* __restrict__ Wdec,
    unsigned short* __restrict__ W2bf, float* __restrict__ bias_enc,
    float* __restrict__ dbias, float* __restrict__ b12) {
  const long N0 = (long)G4_ * KCAT_;
  const long N1 = (long)G4_ * H_;
  const long N2 = (long)U_ * H_;
  const long NT = N0 + N1 + N2 + G4_ + G4_ + U_;
  for (long i = (long)blockIdx.x * blockDim.x + threadIdx.x; i < NT;
       i += (long)gridDim.x * blockDim.x) {
    long j = i;
    if (j < N0) {                       // fused encoder weight [Whh | Wih], bf16
      int g = (int)(j / KCAT_), c = (int)(j % KCAT_);
      float v = (c < H_) ? enc_Whh[(size_t)g * H_ + c]
                         : enc_Wih[(size_t)g * F_ + (c - H_)];
      Wenc[j] = f2bf(v);
    } else if ((j -= N0) < N1) {        // dec_Wih cols 0..511 (y_t==0 -> col 512 dropped)
      int g = (int)(j / H_), k = (int)(j % H_);
      Wdec[j] = f2bf(dec_Wih[(size_t)g * (H_ + 1) + k]);
    } else if ((j -= N1) < N2) {
      W2bf[j] = f2bf(att_W2[j]);
    } else if ((j -= N2) < G4_) {
      bias_enc[j] = enc_bih[j] + enc_bhh[j];
    } else if ((j -= G4_) < G4_) {
      dbias[j] = dec_bih[j] + dec_bhh[j];
    } else {
      j -= G4_;
      b12[j] = att_b1[j] + att_b2[j];
    }
  }
}

// ---- persistent encoder: 512 LSTM steps, 32 WGs, grid barrier per step ----
__global__ __launch_bounds__(256) void enc_kernel(
    const float* __restrict__ x, const unsigned short* __restrict__ Wenc,
    const float* __restrict__ bias, unsigned short* __restrict__ hx,
    float* __restrict__ encf, unsigned short* __restrict__ encb,
    float* __restrict__ h_att, unsigned* __restrict__ sy) {
  __shared__ unsigned short bufA[2][64][KCH_]; // double-buffered activation chunks
  __shared__ float pre_lds[4][64][16];         // gate pre-activations via LDS
  __shared__ float c_lds[64][16];              // persistent cell state
  const int wg = blockIdx.x, tid = threadIdx.x;
  const int wave = tid >> 5, lane = tid & 31;
  const int r = lane & 15, hf = lane >> 4;
  const int jc = wg * 16;                      // this WG owns h columns jc..jc+15
  const int q = wave >> 1, mp = wave & 1;
  const int m0 = mp * 32, m1 = m0 + 16;

  // init: hx[0] = [h=0 | x_t0], c = 0
  for (int i = wg * 256 + tid; i < B_ * KCAT_; i += NWG_ * 256) {
    int b = i / KCAT_, k = i - b * KCAT_;
    hx[i] = (k < H_) ? (unsigned short)0
                     : f2bf(x[((size_t)b * S_) * F_ + (k - H_)]);
  }
  for (int i = tid; i < 64 * 16; i += 256) (&c_lds[0][0])[i] = 0.0f;
  gsync(sy, sy + 1);

  for (int t = 0; t < S_; ++t) {
    const unsigned short* hxr = hx + (size_t)(t & 1) * B_ * KCAT_;
    unsigned short* hxw = hx + (size_t)((t + 1) & 1) * B_ * KCAT_;
    v8f acc0 = {0.f,0.f,0.f,0.f,0.f,0.f,0.f,0.f};
    v8f acc1 = {0.f,0.f,0.f,0.f,0.f,0.f,0.f,0.f};

    stage_chunk(bufA[0], hxr, 0, tid);
    stage_wait(tid);
    for (int kc = 0; kc < KCAT_ / KCH_; ++kc) {
      if (kc + 1 < KCAT_ / KCH_)
        stage_chunk(bufA[(kc + 1) & 1], hxr, kc + 1, tid);  // overlap next chunk
      const unsigned short(*cur)[KCH_] = bufA[kc & 1];
      #pragma unroll
      for (int ki = 0; ki < KCH_ / 32; ++ki) {
        const int k0 = kc * KCH_ + ki * 32;   // global K (for B)
        const int kl = ki * 32 + hf * 8;      // local K in LDS tile
        __builtin_prefetch(Wenc + (size_t)(q * H_ + jc + r) * KCAT_ + (k0 + 32), 0, 3);
        v16bf bf = load_frag(Wenc, KCAT_, q * H_ + jc, k0);  // B: W[n,k] from L2
        Frag a0, a1;                                         // A from LDS
        a0.q[0] = *(const uint4*)&cur[m0 + r][kl];
        a0.q[1] = *(const uint4*)&cur[m0 + r][kl + 16];
        a1.q[0] = *(const uint4*)&cur[m1 + r][kl];
        a1.q[1] = *(const uint4*)&cur[m1 + r][kl + 16];
        acc0 = wmma_bf16(a0.v, bf, acc0);
        acc1 = wmma_bf16(a1.v, bf, acc1);
      }
      stage_wait(tid);   // next chunk landed; also fences LDS reads of cur
    }

    #pragma unroll
    for (int jv = 0; jv < 8; ++jv) {
      pre_lds[q][m0 + hf * 8 + jv][r] = acc0[jv];
      pre_lds[q][m1 + hf * 8 + jv][r] = acc1[jv];
    }
    __syncthreads();
    for (int e = tid; e < 64 * 16; e += 256) {
      int m = e >> 4, n = e & 15, col = jc + n;
      float iv = sigf(pre_lds[0][m][n] + bias[col]);
      float fv = sigf(pre_lds[1][m][n] + bias[H_ + col]);
      float gv = tanhf(pre_lds[2][m][n] + bias[2 * H_ + col]);
      float ov = sigf(pre_lds[3][m][n] + bias[3 * H_ + col]);
      float c = fv * c_lds[m][n] + iv * gv;
      c_lds[m][n] = c;
      float h = ov * tanhf(c);
      size_t eo = ((size_t)m * S_ + t) * H_ + col;
      encf[eo] = h;
      encb[eo] = f2bf(h);
      hxw[(size_t)m * KCAT_ + col] = f2bf(h);
      if (t == S_ - 1) h_att[(size_t)m * H_ + col] = h;
    }
    if (t + 1 < S_ && tid < 128) {       // stage x[:,t+1,:] into the write buffer
      int b = tid >> 1, fc = 2 * wg + (tid & 1);
      hxw[(size_t)b * KCAT_ + H_ + fc] = f2bf(x[((size_t)b * S_ + (t + 1)) * F_ + fc]);
    }
    gsync(sy, sy + 1);
  }
}

// ---- w2_enc = enc_out @ W2.T : M=32768, N=128, K=512 ----
__global__ __launch_bounds__(256) void w2_kernel(const unsigned short* __restrict__ encb,
                                                 const unsigned short* __restrict__ W2bf,
                                                 float* __restrict__ w2enc) {
  const int tid = threadIdx.x, wave = tid >> 5, lane = tid & 31;
  const int r = lane & 15, hf = lane >> 4;
  const int m0 = blockIdx.x * 16, n0 = wave * 16;
  v8f acc = {0.f,0.f,0.f,0.f,0.f,0.f,0.f,0.f};
  for (int kk = 0; kk < H_ / 32; ++kk) {
    v16bf a = load_frag(encb, H_, m0, kk * 32);
    v16bf b = load_frag(W2bf, H_, n0, kk * 32);
    acc = wmma_bf16(a, b, acc);
  }
  #pragma unroll
  for (int jv = 0; jv < 8; ++jv)
    w2enc[(size_t)(m0 + hf * 8 + jv) * U_ + n0 + r] = acc[jv];
}

// ---- Bahdanau scores + softmax, one block per batch ----
__global__ __launch_bounds__(256) void attn_kernel(
    const float* __restrict__ h_att, const float* __restrict__ W1,
    const float* __restrict__ b12, const float* __restrict__ V,
    const float* __restrict__ bV, const float* __restrict__ w2enc,
    float* __restrict__ attn) {
  const int b = blockIdx.x, tid = threadIdx.x;
  __shared__ float w1h[U_];
  __shared__ float red[256];
  __shared__ float sc[S_];
  {
    int u = tid & 127, kh = tid >> 7;
    float p = 0.f;
    const float* hb = h_att + (size_t)b * H_ + kh * 256;
    const float* wb = W1 + (size_t)u * H_ + kh * 256;
    for (int k = 0; k < 256; ++k) p += hb[k] * wb[k];
    red[tid] = p;
  }
  __syncthreads();
  if (tid < 128) w1h[tid] = red[tid] + red[tid + 128] + b12[tid];
  __syncthreads();
  float lmax = -1e30f;
  for (int s = tid; s < S_; s += 256) {
    const float* w2 = w2enc + ((size_t)b * S_ + s) * U_;
    float a = bV[0];
    #pragma unroll 4
    for (int u = 0; u < U_; ++u) a += V[u] * tanhf(w1h[u] + w2[u]);
    sc[s] = a;
    lmax = fmaxf(lmax, a);
  }
  red[tid] = lmax; __syncthreads();
  for (int o = 128; o > 0; o >>= 1) { if (tid < o) red[tid] = fmaxf(red[tid], red[tid + o]); __syncthreads(); }
  float mx = red[0]; __syncthreads();
  float lsum = 0.f;
  for (int s = tid; s < S_; s += 256) { float e = __expf(sc[s] - mx); sc[s] = e; lsum += e; }
  red[tid] = lsum; __syncthreads();
  for (int o = 128; o > 0; o >>= 1) { if (tid < o) red[tid] += red[tid + o]; __syncthreads(); }
  float inv = 1.f / red[0];
  for (int s = tid; s < S_; s += 256) attn[(size_t)b * S_ + s] = sc[s] * inv;
}

// ---- context: ctx[b,h] = sum_s attn[b,s] * enc_out[b,s,h] (L2-resident) ----
__global__ __launch_bounds__(128) void ctx_kernel(const float* __restrict__ attn,
                                                  const float* __restrict__ encf,
                                                  float* __restrict__ ctxf,
                                                  unsigned short* __restrict__ ctxb) {
  const int b = blockIdx.x >> 2;
  const int hc = (blockIdx.x & 3) * 128 + threadIdx.x;
  const float* eb = encf + (size_t)b * S_ * H_ + hc;
  const float* ab = attn + (size_t)b * S_;
  float a = 0.f;
  for (int s = 0; s < S_; ++s) a += ab[s] * eb[(size_t)s * H_];
  ctxf[(size_t)b * H_ + hc] = a;
  ctxb[(size_t)b * H_ + hc] = f2bf(a);
}

// ---- decoder LSTM cell from zero state: pre = ctx @ Wdec.T, fused gates ----
__global__ __launch_bounds__(256) void dec_kernel(const unsigned short* __restrict__ ctxb,
                                                  const unsigned short* __restrict__ Wdec,
                                                  const float* __restrict__ dbias,
                                                  float* __restrict__ h_att) {
  __shared__ float pre_lds[4][64][16];
  const int wg = blockIdx.x, tid = threadIdx.x;
  const int wave = tid >> 5, lane = tid & 31;
  const int r = lane & 15, hf = lane >> 4;
  const int jc = wg * 16, q = wave >> 1, mp = wave & 1;
  const int m0 = mp * 32, m1 = m0 + 16;
  v8f acc0 = {0.f,0.f,0.f,0.f,0.f,0.f,0.f,0.f};
  v8f acc1 = {0.f,0.f,0.f,0.f,0.f,0.f,0.f,0.f};
  for (int kk = 0; kk < H_ / 32; ++kk) {
    v16bf bf = load_frag(Wdec, H_, q * H_ + jc, kk * 32);
    v16bf a0 = load_frag(ctxb, H_, m0, kk * 32);
    v16bf a1 = load_frag(ctxb, H_, m1, kk * 32);
    acc0 = wmma_bf16(a0, bf, acc0);
    acc1 = wmma_bf16(a1, bf, acc1);
  }
  #pragma unroll
  for (int jv = 0; jv < 8; ++jv) {
    pre_lds[q][m0 + hf * 8 + jv][r] = acc0[jv];
    pre_lds[q][m1 + hf * 8 + jv][r] = acc1[jv];
  }
  __syncthreads();
  for (int e = tid; e < 64 * 16; e += 256) {
    int m = e >> 4, n = e & 15, col = jc + n;
    float iv = sigf(pre_lds[0][m][n] + dbias[col]);              // f gate * c0 == 0
    float gv = tanhf(pre_lds[2][m][n] + dbias[2 * H_ + col]);
    float ov = sigf(pre_lds[3][m][n] + dbias[3 * H_ + col]);
    float c = iv * gv;
    h_att[(size_t)m * H_ + col] = ov * tanhf(c);
  }
}

// ---- output projection: out[b,t] = h . out_W + out_b ----
__global__ __launch_bounds__(256) void out_kernel(const float* __restrict__ h_att,
                                                  const float* __restrict__ outW,
                                                  const float* __restrict__ outb,
                                                  float* __restrict__ out, int t) {
  const int b = blockIdx.x, tid = threadIdx.x;
  __shared__ float red[256];
  float p = 0.f;
  for (int k = tid; k < H_; k += 256) p += h_att[(size_t)b * H_ + k] * outW[k];
  red[tid] = p; __syncthreads();
  for (int o = 128; o > 0; o >>= 1) { if (tid < o) red[tid] += red[tid + o]; __syncthreads(); }
  if (tid == 0) out[(size_t)b * TOUT_ + t] = red[0] + outb[0];
}

extern "C" void kernel_launch(void* const* d_in, const int* in_sizes, int n_in,
                              void* d_out, int out_size, void* d_ws, size_t ws_size,
                              hipStream_t stream) {
  const float* x       = (const float*)d_in[0];
  const float* enc_Wih = (const float*)d_in[1];
  const float* enc_Whh = (const float*)d_in[2];
  const float* enc_bih = (const float*)d_in[3];
  const float* enc_bhh = (const float*)d_in[4];
  const float* att_W1  = (const float*)d_in[5];
  const float* att_b1  = (const float*)d_in[6];
  const float* att_W2  = (const float*)d_in[7];
  const float* att_b2  = (const float*)d_in[8];
  const float* att_V   = (const float*)d_in[9];
  const float* att_bV  = (const float*)d_in[10];
  const float* dec_Wih = (const float*)d_in[11];
  const float* dec_bih = (const float*)d_in[13];
  const float* dec_bhh = (const float*)d_in[14];
  const float* out_W   = (const float*)d_in[15];
  const float* out_b   = (const float*)d_in[16];
  float* out = (float*)d_out;

  char* ws = (char*)d_ws;
  size_t off = 0;
  auto take = [&](size_t bytes) -> char* {
    char* p = ws + off;
    off = (off + bytes + 255) & ~(size_t)255;
    return p;
  };
  unsigned* syncp        = (unsigned*)take(256);
  float* bias_enc        = (float*)take((size_t)G4_ * 4);
  float* dbias           = (float*)take((size_t)G4_ * 4);
  float* b12             = (float*)take((size_t)U_ * 4);
  unsigned short* Wenc   = (unsigned short*)take((size_t)G4_ * KCAT_ * 2);
  unsigned short* Wdec   = (unsigned short*)take((size_t)G4_ * H_ * 2);
  unsigned short* W2bf   = (unsigned short*)take((size_t)U_ * H_ * 2);
  unsigned short* hx     = (unsigned short*)take((size_t)2 * B_ * KCAT_ * 2);
  float* h_att           = (float*)take((size_t)B_ * H_ * 4);
  float* ctxf            = (float*)take((size_t)B_ * H_ * 4);
  unsigned short* ctxb   = (unsigned short*)take((size_t)B_ * H_ * 2);
  float* attn            = (float*)take((size_t)B_ * S_ * 4);
  float* w2enc           = (float*)take((size_t)B_ * S_ * U_ * 4);
  unsigned short* encb   = (unsigned short*)take((size_t)B_ * S_ * H_ * 2);
  float* encf            = (float*)take((size_t)B_ * S_ * H_ * 4);

  (void)hipMemsetAsync(syncp, 0, 256, stream);
  prep_kernel<<<2048, 256, 0, stream>>>(enc_Wih, enc_Whh, enc_bih, enc_bhh,
                                        att_b1, att_b2, att_W2, dec_Wih,
                                        dec_bih, dec_bhh, Wenc, Wdec, W2bf,
                                        bias_enc, dbias, b12);
  enc_kernel<<<NWG_, 256, 0, stream>>>(x, Wenc, bias_enc, hx, encf, encb, h_att, syncp);
  w2_kernel<<<(B_ * S_) / 16, 256, 0, stream>>>(encb, W2bf, w2enc);
  for (int t = 0; t < TOUT_; ++t) {
    attn_kernel<<<B_, 256, 0, stream>>>(h_att, att_W1, b12, att_V, att_bV, w2enc, attn);
    ctx_kernel<<<B_ * 4, 128, 0, stream>>>(attn, encf, ctxf, ctxb);
    dec_kernel<<<NWG_, 256, 0, stream>>>(ctxb, Wdec, dbias, h_att);
    out_kernel<<<B_, 256, 0, stream>>>(h_att, out_W, out_b, out, t);
  }
  (void)in_sizes; (void)n_in; (void)out_size; (void)ws_size;
}
